// CompetitionHGT_8478265442687
// MI455X (gfx1250) — matrix operand order repaired and checked by reading, not compiled, not run on hardware
//
#include <hip/hip_runtime.h>
#include <hip/hip_bf16.h>
#include <limits.h>

// ---------------- constants (match reference) ----------------
// NB, NC, NTOT are all multiples of 16 -> WMMA GEMM needs no row guards.
static const int NB = 50000, NC = 10000;
static const int EBB = 400000, EBC = 200000, ECB = 200000;
static const int IN_CH = 128, HID = 128, OUT_CH = 64;
static const int NTOT = NB + NC;

typedef __attribute__((ext_vector_type(16))) _Float16 v16h;
typedef __attribute__((ext_vector_type(8)))  _Float16 v8h;
typedef __attribute__((ext_vector_type(8)))  float    v8f;

// ---------------- small utility kernels ----------------
__global__ void k_fill_f32(float* p, long n, float v) {
    long i = (long)blockIdx.x * blockDim.x + threadIdx.x;
    if (i < n) p[i] = v;
}
__global__ void k_fill_i32(int* p, long n, int v) {
    long i = (long)blockIdx.x * blockDim.x + threadIdx.x;
    if (i < n) p[i] = v;
}
__global__ void k_cast_f32_f16(const float* __restrict__ s, _Float16* __restrict__ d, long n) {
    long i = (long)blockIdx.x * blockDim.x + threadIdx.x;
    if (i < n) d[i] = (_Float16)s[i];
}

// Pack a row-major f32 weight [K, N] (row stride ld) into the per-lane WMMA B-fragment
// layout, f16.  dst index = (((tn*ksteps + ks)*32) + lane)*16 + j  where the lane's 16
// halfs are contiguous:  B[ks*32 + (lane>=16?16:0) + j][tn*16 + (lane&15)].
__global__ void k_pack_b(const float* __restrict__ src, int ld, int K, int N,
                         _Float16* __restrict__ dst) {
    long n = (long)K * N;
    long i = (long)blockIdx.x * blockDim.x + threadIdx.x;
    if (i >= n) return;
    int ksteps = K >> 5;
    int j    = (int)(i & 15);
    long t1  = i >> 4;
    int lane = (int)(t1 & 31);
    long t2  = t1 >> 5;
    int ks   = (int)(t2 % ksteps);
    int tn   = (int)(t2 / ksteps);
    int k    = ks * 32 + ((lane >= 16) ? 16 : 0) + j;
    int col  = tn * 16 + (lane & 15);
    dst[i] = (_Float16)src[(size_t)k * ld + col];
}

// Fuse per-relation head-blockdiag transform into projection weight (fp32, exact):
// out[r, h*D+j] = sum_d W[r, h*D+d] * rel[h, d, j]
__global__ void k_fuse_weight(const float* __restrict__ W, int ldw,
                              const float* __restrict__ rel,
                              float* __restrict__ out, int rows, int oc, int D) {
    long i = (long)blockIdx.x * blockDim.x + threadIdx.x;
    long n = (long)rows * oc;
    if (i >= n) return;
    int r = (int)(i / oc), c = (int)(i % oc);
    int h = c / D, j = c % D;
    const float* wrow = W + (size_t)r * ldw + (size_t)h * D;
    const float* rj   = rel + (size_t)h * D * D + j;
    float acc = 0.f;
    for (int d = 0; d < D; ++d) acc += wrow[d] * rj[(size_t)d * D];
    out[i] = acc;
}

// ---------------- WMMA GEMM:  C[M,N](f32) = A[M,K](f16) * Bp(packed f16) + bias[N] ----
// Row-major A (lda, 16B-aligned rows), packed B, row-major C (ldc).
// Requirements: M % 16 == 0, N % 32 == 0, K % 32 == 0.
// One wave computes a 16x32 tile (two accumulators sharing the A fragment).
__global__ void k_wmma_gemm(const _Float16* __restrict__ A, int lda,
                            const _Float16* __restrict__ Bp,
                            const float* __restrict__ bias,
                            float* __restrict__ C, int ldc,
                            int M, int K, int tilesN2) {
    int lane = threadIdx.x & 31;
    int wave = threadIdx.x >> 5;
    long tile = (long)blockIdx.x * 4 + wave;
    long totalTiles = (long)(M >> 4) * tilesN2;
    if (tile >= totalTiles) return;
    int tm  = (int)(tile / tilesN2);
    int tn2 = (int)(tile % tilesN2);
    int lh  = lane & 15;
    bool hi = lane >= 16;
    int ksteps = K >> 5;

    const _Float16* Arow = A + (size_t)(tm * 16 + lh) * lda + (hi ? 16 : 0);
    const _Float16* bp0 = Bp + ((size_t)(2 * tn2)     * ksteps * 32 + lane) * 16;
    const _Float16* bp1 = Bp + ((size_t)(2 * tn2 + 1) * ksteps * 32 + lane) * 16;

    v8f acc0 = {0.f, 0.f, 0.f, 0.f, 0.f, 0.f, 0.f, 0.f};
    v8f acc1 = {0.f, 0.f, 0.f, 0.f, 0.f, 0.f, 0.f, 0.f};
    for (int ks = 0; ks < ksteps; ++ks) {
        // A fragment: two contiguous 8-half (16 B) chunks per lane.
        v8h a0 = *(const v8h*)(Arow + ks * 32);
        v8h a1 = *(const v8h*)(Arow + ks * 32 + 8);
        v16h av = __builtin_shufflevector(a0, a1, 0, 1, 2, 3, 4, 5, 6, 7,
                                          8, 9, 10, 11, 12, 13, 14, 15);
        // Packed B fragments: lane's 16 halfs are contiguous (32 B).
        v16h bv0 = *(const v16h*)(bp0 + (size_t)ks * 512);
        v16h bv1 = *(const v16h*)(bp1 + (size_t)ks * 512);
        acc0 = __builtin_amdgcn_wmma_f32_16x16x32_f16(false, av, false, bv0,
                                                      (short)0, acc0, false, false);
        acc1 = __builtin_amdgcn_wmma_f32_16x16x32_f16(false, av, false, bv1,
                                                      (short)0, acc1, false, false);
    }
    int colBase = tn2 * 32 + lh;
    float b0 = bias[colBase], b1 = bias[colBase + 16];
    float* crow = C + (size_t)(tm * 16 + (hi ? 8 : 0)) * ldc + colBase;
#pragma unroll
    for (int i = 0; i < 8; ++i) {
        crow[(size_t)i * ldc]      = acc0[i] + b0;
        crow[(size_t)i * ldc + 16] = acc1[i] + b1;
    }
}

// ---------------- edge-stage kernels (wave per edge, wave32) ----------------
__device__ inline int f_ordered(float f) {
    int i = __float_as_int(f);
    return (i >= 0) ? i : (i ^ 0x7fffffff);
}

__global__ void k_edge_logits(const int* __restrict__ src, const int* __restrict__ dst, int E,
                              const float* __restrict__ Q, const float* __restrict__ Krel,
                              int dstOff, const float* __restrict__ prel, float scale,
                              float* __restrict__ logits, int* __restrict__ segmax,
                              int H, int D, long base) {
    int lane = threadIdx.x & 31;
    int e = blockIdx.x * (blockDim.x >> 5) + (threadIdx.x >> 5);
    if (e >= E) return;
    int s  = src[e];
    int dg = dst[e] + dstOff;
    int HD = H * D;
    const float* q = Q    + (size_t)dg * HD;
    const float* k = Krel + (size_t)s  * HD;
    float p[4] = {0.f, 0.f, 0.f, 0.f};
    for (int c = lane; c < HD; c += 32) p[c / D] += q[c] * k[c];
#pragma unroll
    for (int off = 16; off >= 1; off >>= 1) {
#pragma unroll
        for (int h = 0; h < 4; ++h) p[h] += __shfl_xor(p[h], off, 32);
    }
    if (lane < H) {
        float l = p[lane] * prel[lane] * scale;
        logits[(base + e) * H + lane] = l;
        atomicMax(&segmax[(size_t)dg * H + lane], f_ordered(l));
    }
}

// decode ordered-int max in place -> float; non-finite (incl. empty-segment NaN) -> 0
__global__ void k_max_finalize(int* __restrict__ p, long n) {
    long i = (long)blockIdx.x * blockDim.x + threadIdx.x;
    if (i >= n) return;
    int v = p[i];
    float m = __int_as_float(v >= 0 ? v : (v ^ 0x7fffffff));
    if (!__builtin_isfinite(m)) m = 0.f;
    ((float*)p)[i] = m;
}

__global__ void k_edge_exp(const int* __restrict__ dst, int E, int dstOff,
                           const float* __restrict__ m, float* __restrict__ logits,
                           float* __restrict__ segsum, int H, long base) {
    long t = (long)blockIdx.x * blockDim.x + threadIdx.x;
    if (t >= (long)E * H) return;
    int e = (int)(t / H), h = (int)(t % H);
    int dg = dst[e] + dstOff;
    long li = (base + e) * H + h;
    float ex = expf(logits[li] - m[(size_t)dg * H + h]);
    logits[li] = ex;
    atomicAdd(&segsum[(size_t)dg * H + h], ex);
}

__global__ void k_inv_sum(float* __restrict__ s, long n) {
    long i = (long)blockIdx.x * blockDim.x + threadIdx.x;
    if (i < n) s[i] = 1.f / (s[i] + 1e-16f);
}

__global__ void k_edge_aggregate(const int* __restrict__ src, const int* __restrict__ dst, int E,
                                 const float* __restrict__ Vrel, const float* __restrict__ ebuf,
                                 const float* __restrict__ rsum, float* __restrict__ agg,
                                 int dstOff, int H, int D, long base) {
    int lane = threadIdx.x & 31;
    int e = blockIdx.x * (blockDim.x >> 5) + (threadIdx.x >> 5);
    if (e >= E) return;
    int s  = src[e];
    int dg = dst[e] + dstOff;
    int HD = H * D;
    float att[4] = {0.f, 0.f, 0.f, 0.f};
    for (int h = 0; h < H; ++h)
        att[h] = ebuf[(base + e) * H + h] * rsum[(size_t)dg * H + h];
    const float* v = Vrel + (size_t)s * HD;
    float* a = agg + (size_t)dg * HD;
    for (int c = lane; c < HD; c += 32)
        atomicAdd(&a[c], att[c / D] * v[c]);
}

// ---------------- pointwise tails ----------------
__global__ void k_gelu_cast(const float* __restrict__ x, _Float16* __restrict__ o, long n) {
    long i = (long)blockIdx.x * blockDim.x + threadIdx.x;
    if (i >= n) return;
    float v = x[i];
    o[i] = (_Float16)(0.5f * v * (1.f + erff(v * 0.70710678118654752f)));
}
// layer-1 tail: gated skip + relu + cast-to-f16 (next layer input)
__global__ void k_skip_relu_cast(const float* __restrict__ a, const float* __restrict__ x,
                                 const float* __restrict__ skipArr, int t, long n,
                                 _Float16* __restrict__ o) {
    long i = (long)blockIdx.x * blockDim.x + threadIdx.x;
    if (i >= n) return;
    float beta = 1.f / (1.f + expf(-skipArr[t]));
    float v = beta * a[i] + (1.f - beta) * x[i];
    o[i] = (_Float16)(v > 0.f ? v : 0.f);
}

// ---------------- host-side orchestration ----------------
static inline dim3 g1(long n, int b) { return dim3((unsigned)((n + b - 1) / b)); }

static void gemm(hipStream_t st, const _Float16* A, int lda, const _Float16* Bp,
                 const float* bias, float* C, int ldc, int M, int N, int K) {
    int tilesN2 = N / 32;
    long tiles = (long)(M / 16) * tilesN2;
    k_wmma_gemm<<<g1(tiles, 4), 128, 0, st>>>(A, lda, Bp, bias, C, ldc, M, K, tilesN2);
}

struct Scratch {
    _Float16* in16;            // [NTOT, in_ch] f16 (layer input, reused between layers)
    float* Q;                  // [NTOT, oc]
    float* KV[3];              // per relation [Nsrc, oc] (K first, reused for V)
    float* logits;             // [Etot, H]
    int*   segmax;             // [NTOT, H] (ordered int -> float in place)
    float* segsum;             // [NTOT, H]
    float* agg;                // [NTOT, oc]
    _Float16* g16;             // [NTOT, oc]
    float* outa;               // [NTOT, oc] (layer-1 pre-skip output)
    _Float16 *wq16[2], *wk16[3], *wv16[3], *wout16[2];  // packed B fragments
    float* wtmp;               // [in_ch, oc] fp32 fuse temp
    float *bkf[3], *bvf[3];    // fused biases [oc]
};

static void run_layer(hipStream_t st, const Scratch& w,
                      const float* Wkqv, const float* bkqv, const float* Wout,
                      const float* bout, const float* skip,
                      const float* arel, const float* mrel, const float* prel,
                      int H, int D, int in_ch, int oc,
                      const int* const* esrc, const int* const* edst, const int* E,
                      const long* ebase, const int* srcType, const int* dstOff,
                      const float* xb_f32, const float* xc_f32,   // layer-1 skip inputs (else null)
                      float* final_out) {                         // layer-2 output (else null)
    const int off[2] = {0, NB};
    const int Nt[2]  = {NB, NC};
    const long row3  = (long)in_ch * 3 * oc;
    const float scale = 1.f / sqrtf((float)D);

    // --- weight prep: fuse (fp32, exact) then pack into WMMA B-fragment layout ---
    for (int t = 0; t < 2; ++t) {
        k_pack_b<<<g1((long)in_ch * oc, 256), 256, 0, st>>>(
            Wkqv + (size_t)t * row3 + oc, 3 * oc, in_ch, oc, w.wq16[t]);
        k_pack_b<<<g1((long)oc * oc, 256), 256, 0, st>>>(
            Wout + (size_t)t * oc * oc, oc, oc, oc, w.wout16[t]);
    }
    for (int r = 0; r < 3; ++r) {
        int stp = srcType[r];
        const float* Wbase = Wkqv + (size_t)stp * row3;
        const float* bbase = bkqv + (size_t)stp * 3 * oc;
        // K: columns [0, oc)
        k_fuse_weight<<<g1((long)in_ch * oc, 256), 256, 0, st>>>(
            Wbase, 3 * oc, arel + (size_t)r * H * D * D, w.wtmp, in_ch, oc, D);
        k_pack_b<<<g1((long)in_ch * oc, 256), 256, 0, st>>>(
            w.wtmp, oc, in_ch, oc, w.wk16[r]);
        k_fuse_weight<<<g1(oc, 128), 128, 0, st>>>(
            bbase, 3 * oc, arel + (size_t)r * H * D * D, w.bkf[r], 1, oc, D);
        // V: columns [2*oc, 3*oc)
        k_fuse_weight<<<g1((long)in_ch * oc, 256), 256, 0, st>>>(
            Wbase + 2 * oc, 3 * oc, mrel + (size_t)r * H * D * D, w.wtmp, in_ch, oc, D);
        k_pack_b<<<g1((long)in_ch * oc, 256), 256, 0, st>>>(
            w.wtmp, oc, in_ch, oc, w.wv16[r]);
        k_fuse_weight<<<g1(oc, 128), 128, 0, st>>>(
            bbase + 2 * oc, 3 * oc, mrel + (size_t)r * H * D * D, w.bvf[r], 1, oc, D);
    }

    // --- Q projections ---
    for (int t = 0; t < 2; ++t)
        gemm(st, w.in16 + (size_t)off[t] * in_ch, in_ch, w.wq16[t],
             bkqv + (size_t)t * 3 * oc + oc, w.Q + (size_t)off[t] * oc, oc,
             Nt[t], oc, in_ch);
    // --- fused K projections ---
    for (int r = 0; r < 3; ++r) {
        int stp = srcType[r];
        gemm(st, w.in16 + (size_t)off[stp] * in_ch, in_ch, w.wk16[r],
             w.bkf[r], w.KV[r], oc, Nt[stp], oc, in_ch);
    }

    // --- logits + segment max ---
    k_fill_i32<<<g1((long)NTOT * H, 256), 256, 0, st>>>(w.segmax, (long)NTOT * H, INT_MIN);
    for (int r = 0; r < 3; ++r)
        k_edge_logits<<<g1(E[r], 8), 256, 0, st>>>(
            esrc[r], edst[r], E[r], w.Q, w.KV[r], dstOff[r], prel + (size_t)r * H,
            scale, w.logits, w.segmax, H, D, ebase[r]);
    k_max_finalize<<<g1((long)NTOT * H, 256), 256, 0, st>>>(w.segmax, (long)NTOT * H);
    k_fill_f32<<<g1((long)NTOT * H, 256), 256, 0, st>>>(w.segsum, (long)NTOT * H, 0.f);
    // --- exp + segment sum ---
    for (int r = 0; r < 3; ++r)
        k_edge_exp<<<g1((long)E[r] * H, 256), 256, 0, st>>>(
            edst[r], E[r], dstOff[r], (const float*)w.segmax, w.logits, w.segsum, H, ebase[r]);
    k_inv_sum<<<g1((long)NTOT * H, 256), 256, 0, st>>>(w.segsum, (long)NTOT * H);

    // --- fused V projections (reuse KV buffers) ---
    for (int r = 0; r < 3; ++r) {
        int stp = srcType[r];
        gemm(st, w.in16 + (size_t)off[stp] * in_ch, in_ch, w.wv16[r],
             w.bvf[r], w.KV[r], oc, Nt[stp], oc, in_ch);
    }
    // --- weighted scatter-aggregate ---
    k_fill_f32<<<g1((long)NTOT * oc, 256), 256, 0, st>>>(w.agg, (long)NTOT * oc, 0.f);
    for (int r = 0; r < 3; ++r)
        k_edge_aggregate<<<g1(E[r], 8), 256, 0, st>>>(
            esrc[r], edst[r], E[r], w.KV[r], w.logits, w.segsum, w.agg,
            dstOff[r], H, D, ebase[r]);

    // --- gelu -> typed output projection -> (skip|out) ---
    k_gelu_cast<<<g1((long)NTOT * oc, 256), 256, 0, st>>>(w.agg, w.g16, (long)NTOT * oc);
    for (int t = 0; t < 2; ++t) {
        float* C = final_out ? (final_out + (size_t)off[t] * oc)
                             : (w.outa + (size_t)off[t] * oc);
        gemm(st, w.g16 + (size_t)off[t] * oc, oc, w.wout16[t],
             bout + (size_t)t * oc, C, oc, Nt[t], oc, oc);
    }
    if (!final_out) {  // layer 1: gated skip (dims match) + relu + f16 cast for next layer
        const float* xs[2] = {xb_f32, xc_f32};
        for (int t = 0; t < 2; ++t)
            k_skip_relu_cast<<<g1((long)Nt[t] * oc, 256), 256, 0, st>>>(
                w.outa + (size_t)off[t] * oc, xs[t], skip, t,
                (long)Nt[t] * oc, w.in16 + (size_t)off[t] * oc);
    }
}

extern "C" void kernel_launch(void* const* d_in, const int* in_sizes, int n_in,
                              void* d_out, int out_size, void* d_ws, size_t ws_size,
                              hipStream_t stream) {
    const float* x_b = (const float*)d_in[0];
    const float* x_c = (const float*)d_in[1];
    const int* esrc[3] = {(const int*)d_in[2], (const int*)d_in[4], (const int*)d_in[6]};
    const int* edst[3] = {(const int*)d_in[3], (const int*)d_in[5], (const int*)d_in[7]};
    const float* l1W  = (const float*)d_in[8],  *l1b  = (const float*)d_in[9];
    const float* l1Wo = (const float*)d_in[10], *l1bo = (const float*)d_in[11];
    const float* l1sk = (const float*)d_in[12];
    const float* l1a  = (const float*)d_in[13], *l1m  = (const float*)d_in[14];
    const float* l1p  = (const float*)d_in[15];
    const float* l2W  = (const float*)d_in[16], *l2b  = (const float*)d_in[17];
    const float* l2Wo = (const float*)d_in[18], *l2bo = (const float*)d_in[19];
    const float* l2sk = (const float*)d_in[20];
    const float* l2a  = (const float*)d_in[21], *l2m  = (const float*)d_in[22];
    const float* l2p  = (const float*)d_in[23];
    float* out = (float*)d_out;

    const int E[3]       = {EBB, EBC, ECB};
    const long ebase[3]  = {0, EBB, (long)EBB + EBC};
    const int srcType[3] = {0, 0, 1};
    const int dstOff[3]  = {0, NB, 0};

    // --- bump-allocate workspace (sized for the larger layer: oc=128, H=4) ---
    char* p = (char*)d_ws;
    auto alloc = [&](size_t bytes) -> void* {
        void* r = (void*)p;
        p += (bytes + 255) & ~(size_t)255;
        return r;
    };
    Scratch w;
    w.in16   = (_Float16*)alloc((size_t)NTOT * 128 * sizeof(_Float16));
    w.Q      = (float*)alloc((size_t)NTOT * 128 * sizeof(float));
    w.KV[0]  = (float*)alloc((size_t)NB * 128 * sizeof(float));
    w.KV[1]  = (float*)alloc((size_t)NB * 128 * sizeof(float));
    w.KV[2]  = (float*)alloc((size_t)NC * 128 * sizeof(float));
    w.logits = (float*)alloc((size_t)(EBB + EBC + ECB) * 4 * sizeof(float));
    w.segmax = (int*)alloc((size_t)NTOT * 4 * sizeof(int));
    w.segsum = (float*)alloc((size_t)NTOT * 4 * sizeof(float));
    w.agg    = (float*)alloc((size_t)NTOT * 128 * sizeof(float));
    w.g16    = (_Float16*)alloc((size_t)NTOT * 128 * sizeof(_Float16));
    w.outa   = (float*)alloc((size_t)NTOT * 128 * sizeof(float));
    for (int t = 0; t < 2; ++t) {
        w.wq16[t]   = (_Float16*)alloc((size_t)128 * 128 * sizeof(_Float16));
        w.wout16[t] = (_Float16*)alloc((size_t)128 * 128 * sizeof(_Float16));
    }
    for (int r = 0; r < 3; ++r) {
        w.wk16[r] = (_Float16*)alloc((size_t)128 * 128 * sizeof(_Float16));
        w.wv16[r] = (_Float16*)alloc((size_t)128 * 128 * sizeof(_Float16));
        w.bkf[r]  = (float*)alloc(128 * sizeof(float));
        w.bvf[r]  = (float*)alloc(128 * sizeof(float));
    }
    w.wtmp = (float*)alloc((size_t)128 * 128 * sizeof(float));

    // --- cast fp32 inputs to f16 unified node-feature buffer ---
    k_cast_f32_f16<<<g1((long)NB * IN_CH, 256), 256, 0, stream>>>(x_b, w.in16, (long)NB * IN_CH);
    k_cast_f32_f16<<<g1((long)NC * IN_CH, 256), 256, 0, stream>>>(
        x_c, w.in16 + (size_t)NB * IN_CH, (long)NC * IN_CH);

    // --- layer 1: H=4, D=32, 128 -> 128, gated skip + relu, produces f16 h in w.in16 ---
    run_layer(stream, w, l1W, l1b, l1Wo, l1bo, l1sk, l1a, l1m, l1p,
              4, 32, IN_CH, HID, esrc, edst, E, ebase, srcType, dstOff,
              x_b, x_c, nullptr);

    // --- layer 2: H=1, D=64, 128 -> 64, no skip (dim mismatch), writes d_out directly ---
    run_layer(stream, w, l2W, l2b, l2Wo, l2bo, l2sk, l2a, l2m, l2p,
              1, 64, HID, OUT_CH, esrc, edst, E, ebase, srcType, dstOff,
              nullptr, nullptr, out);

    (void)in_sizes; (void)n_in; (void)out_size; (void)ws_size;
}
